// CascadeHadamardLinear_49787260895357
// MI455X (gfx1250) — compile-verified, matrix operand order checked by reading, not run on hardware
//
#include <hip/hip_runtime.h>
#include <hip/hip_bf16.h>

// ---------------------------------------------------------------------------
// CascadeHadamardLinear for MI455X (gfx1250, wave32, WMMA)
//   B=4, S=2048 -> M=8192 tokens, D_IN=K=4096, D_OUT=N=4096, RANK=32,
//   Hadamard block 128, quant block 16.
// Pipeline:
//   prep_kernel    : sign * x, 128-pt FWHT (butterfly in LDS), NVFP4 fake-quant
//                    -> fp16 xq ; deterministic LoRA projection t = x_rot @ A^T
//   convert_kernel : w, lora_b fp32 -> fp16
//   gemm_kernel    : out = xq @ w^T (+ t @ lora_b^T + bias) via
//                    v_wmma_f32_16x16x32_f16, double-buffered LDS, K-tile 64.
//                    Tile fill uses GLOBAL_LOAD_ASYNC_TO_LDS_B128 (ASYNCcnt,
//                    zero VGPR staging -> no spills); falls back to register
//                    staging if the async builtin is unavailable.
// ---------------------------------------------------------------------------

#define M_TOK 8192
#define DK    4096
#define DN    4096
#define RANK  32
#define HBSZ  128
#define QBSZ  16

typedef __attribute__((ext_vector_type(16))) _Float16 v16h;
typedef __attribute__((ext_vector_type(8)))  _Float16 v8h;
typedef __attribute__((ext_vector_type(8)))  float    v8f;

#if __has_builtin(__builtin_amdgcn_global_load_async_to_lds_b128)
#define HAVE_ASYNC_LDS 1
// exact param type from hipcc diagnostic: generic `int __vector(4) *`
typedef int async_i4 __attribute__((vector_size(16)));
#else
#define HAVE_ASYNC_LDS 0
#endif

__device__ __forceinline__ void wait_async0() {
#if __has_builtin(__builtin_amdgcn_s_wait_asynccnt)
    __builtin_amdgcn_s_wait_asynccnt(0);
#else
    asm volatile("s_wait_asynccnt 0x0" ::: "memory");
#endif
}

// ---------------------------------------------------------------------------
// Kernel 1: rotate + fake-quant + LoRA projection. One 256-thread block/token.
// ---------------------------------------------------------------------------
__global__ __launch_bounds__(256) void prep_kernel(
    const float* __restrict__ x,       // [M_TOK, DK]
    const float* __restrict__ sgn,     // [DK]
    const float* __restrict__ lora_a,  // [RANK, DK]
    _Float16* __restrict__ xq,         // [M_TOK, DK] fp16
    _Float16* __restrict__ tmat)       // [M_TOK, RANK] fp16
{
    __shared__ float row[DK];
    __shared__ float pred[8][RANK];

    const int token = blockIdx.x;
    const int t     = threadIdx.x;
    const float* xrow = x + (size_t)token * DK;

    // ---- load + sign flip (vectorized) ----
    for (int i = t; i < DK / 4; i += 256) {
        float4 v = ((const float4*)xrow)[i];
        float4 s = ((const float4*)sgn)[i];
        row[4 * i + 0] = v.x * s.x;
        row[4 * i + 1] = v.y * s.y;
        row[4 * i + 2] = v.z * s.z;
        row[4 * i + 3] = v.w * s.w;
    }
    __syncthreads();

    // ---- fast Walsh-Hadamard transform per 128-block (Sylvester order) ----
    for (int s = 0; s < 7; ++s) {
        const int stride = 1 << s;
        #pragma unroll
        for (int p = 0; p < 8; ++p) {
            int pg  = t + p * 256;          // 0..2047 pairs
            int blk = pg >> 6;              // 32 Hadamard blocks
            int j   = pg & 63;              // 64 pairs per block
            int i   = ((j / stride) * (2 * stride)) + (j % stride);
            int idx = blk * HBSZ + i;
            float a = row[idx];
            float b = row[idx + stride];
            row[idx]          = a + b;
            row[idx + stride] = a - b;
        }
        __syncthreads();
    }

    const float inv = 0.08838834764831845f;  // 1/sqrt(128)

    // ---- NVFP4 fake-quant: one 16-element block per thread ----
    {
        const int base = t * QBSZ;
        float v[QBSZ];
        float amax = 0.0f;
        #pragma unroll
        for (int i = 0; i < QBSZ; ++i) {
            v[i] = row[base + i] * inv;
            amax = fmaxf(amax, fabsf(v[i]));
        }
        amax = fmaxf(amax, 1e-12f);
        float scale = amax * (1.0f / 6.0f);
        float rs    = 6.0f / amax;

        union { _Float16 h[QBSZ]; uint4 u[2]; } uu;
        #pragma unroll
        for (int i = 0; i < QBSZ; ++i) {
            float an = fabsf(v[i]) * rs;
            float lv;                         // midpoints, ties -> lower level
            if      (an <= 0.25f) lv = 0.0f;
            else if (an <= 0.75f) lv = 0.5f;
            else if (an <= 1.25f) lv = 1.0f;
            else if (an <= 1.75f) lv = 1.5f;
            else if (an <= 2.5f)  lv = 2.0f;
            else if (an <= 3.5f)  lv = 3.0f;
            else if (an <= 5.0f)  lv = 4.0f;
            else                  lv = 6.0f;
            uu.h[i] = (_Float16)copysignf(lv * scale, v[i]);
        }
        uint4* dst = (uint4*)(xq + (size_t)token * DK + base);
        dst[0] = uu.u[0];
        dst[1] = uu.u[1];
    }

    // ---- LoRA projection t[token, r] = sum_d x_rot[d] * lora_a[r, d] ----
    // Deterministic: thread (seg = t>>5, r = t&31) owns a fixed 512-wide slice.
    {
        const int r   = t & 31;
        const int seg = t >> 5;
        const float* ar  = lora_a + (size_t)r * DK + seg * 512;
        const float* rs2 = row + seg * 512;
        float acc = 0.0f;
        #pragma unroll 8
        for (int i = 0; i < 512; ++i) acc = fmaf(rs2[i], ar[i], acc);
        pred[seg][r] = acc * inv;
    }
    __syncthreads();
    if (t < RANK) {
        float acc = 0.0f;
        #pragma unroll
        for (int g = 0; g < 8; ++g) acc += pred[g][t];  // fixed order
        tmat[(size_t)token * RANK + t] = (_Float16)acc;
    }
}

// ---------------------------------------------------------------------------
// Kernel 2: weight / lora_b fp32 -> fp16
// ---------------------------------------------------------------------------
__global__ __launch_bounds__(256) void convert_kernel(
    const float* __restrict__ w, const float* __restrict__ lb,
    _Float16* __restrict__ w16, _Float16* __restrict__ lb16,
    int nw, int nlb)
{
    int idx    = blockIdx.x * 256 + threadIdx.x;
    int stride = gridDim.x * 256;
    for (int i = idx; i < nw;  i += stride) w16[i]  = (_Float16)w[i];
    for (int i = idx; i < nlb; i += stride) lb16[i] = (_Float16)lb[i];
}

// ---------------------------------------------------------------------------
// Kernel 3: WMMA GEMM, double-buffered.
// Block tile 128x128, 8 waves x (32x64), K staged 64 per iteration.
// A = xq [M,K], B = w16 [N,K] (row n of w16 is column n of B = W^T).
// Fragment gather (wave32, 16-bit, per ISA layout tables):
//   lane&15 selects the M (A) / N (B) line, khalf = (lane>=16)*8 selects the
//   K sub-half; each lane packs K = {khalf..khalf+7, khalf+16..khalf+23}.
// Buffer q=(i+1)&1 was last read before the end-of-iter-(i-1) barrier, so
// filling it during iter i with one barrier per iteration is race-free.
// ---------------------------------------------------------------------------
__global__ __launch_bounds__(256) void gemm_kernel(
    const _Float16* __restrict__ A,   // [M, K]   quantized activations
    const _Float16* __restrict__ Bw,  // [N, K]   fp16 weights
    const _Float16* __restrict__ T,   // [M, 32]  LoRA mid
    const _Float16* __restrict__ Lb,  // [N, 32]  fp16 lora_b
    const float*    __restrict__ bias,
    float* __restrict__ C, int M, int N, int K)
{
    __shared__ __align__(16) _Float16 As[2][128 * 64];  // 2 x 16 KB
    __shared__ __align__(16) _Float16 Bs[2][128 * 64];  // 2 x 16 KB

    const int t     = threadIdx.x;
    const int lane  = t & 31;
    const int wave  = t >> 5;          // 0..7
    const int wm    = wave & 3;        // 4 x 32-row strips
    const int wn    = wave >> 2;       // 2 x 64-col strips
    const int mbase = blockIdx.y * 128;
    const int nbase = blockIdx.x * 128;

    const int col16 = lane & 15;
    const int khalf = (lane >> 4) * 8; // 0 or 8

    // per-thread staging slots: chunk c covers row (t+c*256)/8, halves ((..)&7)*8
    const int srow[4] = { (t + 0*256) >> 3, (t + 1*256) >> 3,
                          (t + 2*256) >> 3, (t + 3*256) >> 3 };
    const int soff[4] = { ((t + 0*256) & 7) * 8, ((t + 1*256) & 7) * 8,
                          ((t + 2*256) & 7) * 8, ((t + 3*256) & 7) * 8 };

    v8f zf = {};
    v8f acc[2][4];
    #pragma unroll
    for (int i = 0; i < 2; ++i)
        #pragma unroll
        for (int j = 0; j < 4; ++j) acc[i][j] = zf;

    const int NITER = K / 64;

#if HAVE_ASYNC_LDS
    // ---- async DMA global -> LDS, no VGPR staging (ASYNCcnt-tracked) ----
    #define STAGE_TILE(buf, kk)                                                  \
        do {                                                                     \
            _Pragma("unroll")                                                    \
            for (int c = 0; c < 4; ++c) {                                        \
                __builtin_amdgcn_global_load_async_to_lds_b128(                  \
                    (async_i4*)&A [(size_t)(mbase + srow[c]) * K + (kk) + soff[c]], \
                    (async_i4*)&As[buf][srow[c] * 64 + soff[c]], 0, 0);          \
                __builtin_amdgcn_global_load_async_to_lds_b128(                  \
                    (async_i4*)&Bw[(size_t)(nbase + srow[c]) * K + (kk) + soff[c]], \
                    (async_i4*)&Bs[buf][srow[c] * 64 + soff[c]], 0, 0);          \
            }                                                                    \
        } while (0)

    // prologue: fill buffer 0
    STAGE_TILE(0, 0);
    wait_async0();
    __syncthreads();
#else
    // ---- fallback: register staging ----
    uint4 ra[4], rb[4];
    #pragma unroll
    for (int c = 0; c < 4; ++c) {
        ra[c] = *(const uint4*)&A [(size_t)(mbase + srow[c]) * K + soff[c]];
        rb[c] = *(const uint4*)&Bw[(size_t)(nbase + srow[c]) * K + soff[c]];
    }
    #pragma unroll
    for (int c = 0; c < 4; ++c) {
        *(uint4*)&As[0][srow[c] * 64 + soff[c]] = ra[c];
        *(uint4*)&Bs[0][srow[c] * 64 + soff[c]] = rb[c];
    }
    __syncthreads();
#endif

    for (int i = 0; i < NITER; ++i) {
        const int p = i & 1;

#if HAVE_ASYNC_LDS
        if (i + 1 < NITER) STAGE_TILE((i + 1) & 1, (i + 1) * 64);
#else
        if (i + 1 < NITER) {
            const int kk = (i + 1) * 64;
            #pragma unroll
            for (int c = 0; c < 4; ++c) {
                ra[c] = *(const uint4*)&A [(size_t)(mbase + srow[c]) * K + kk + soff[c]];
                rb[c] = *(const uint4*)&Bw[(size_t)(nbase + srow[c]) * K + kk + soff[c]];
            }
        }
#endif

        // ---- 2 x (8x v_wmma_f32_16x16x32_f16) on current buffer ----
        #pragma unroll
        for (int ks = 0; ks < 2; ++ks) {
            const int kof = ks * 32;
            v16h af[2], bf[4];
            #pragma unroll
            for (int ii = 0; ii < 2; ++ii) {
                int rowm = wm * 32 + ii * 16 + col16;
                v8h lo = *(const v8h*)&As[p][rowm * 64 + kof + khalf];
                v8h hi = *(const v8h*)&As[p][rowm * 64 + kof + khalf + 16];
                #pragma unroll
                for (int e = 0; e < 8; ++e) { af[ii][e] = lo[e]; af[ii][8 + e] = hi[e]; }
            }
            #pragma unroll
            for (int j = 0; j < 4; ++j) {
                int rown = wn * 64 + j * 16 + col16;
                v8h lo = *(const v8h*)&Bs[p][rown * 64 + kof + khalf];
                v8h hi = *(const v8h*)&Bs[p][rown * 64 + kof + khalf + 16];
                #pragma unroll
                for (int e = 0; e < 8; ++e) { bf[j][e] = lo[e]; bf[j][8 + e] = hi[e]; }
            }
            #pragma unroll
            for (int ii = 0; ii < 2; ++ii)
                #pragma unroll
                for (int j = 0; j < 4; ++j)
                    acc[ii][j] = __builtin_amdgcn_wmma_f32_16x16x32_f16(
                        false, af[ii], false, bf[j], (short)0, acc[ii][j], false, false);
        }

#if HAVE_ASYNC_LDS
        wait_async0();           // next tile fully resident in LDS
#else
        if (i + 1 < NITER) {
            const int q = (i + 1) & 1;
            #pragma unroll
            for (int c = 0; c < 4; ++c) {
                *(uint4*)&As[q][srow[c] * 64 + soff[c]] = ra[c];
                *(uint4*)&Bs[q][srow[c] * 64 + soff[c]] = rb[c];
            }
        }
#endif
        __syncthreads();
    }

    // ---- LoRA epilogue: one extra K=32 WMMA per C fragment ----
    {
        v16h tf[2], lf[4];
        #pragma unroll
        for (int i = 0; i < 2; ++i) {
            int rowm = mbase + wm * 32 + i * 16 + col16;
            v8h lo = *(const v8h*)&T[(size_t)rowm * 32 + khalf];
            v8h hi = *(const v8h*)&T[(size_t)rowm * 32 + khalf + 16];
            #pragma unroll
            for (int e = 0; e < 8; ++e) { tf[i][e] = lo[e]; tf[i][8 + e] = hi[e]; }
        }
        #pragma unroll
        for (int j = 0; j < 4; ++j) {
            int rown = nbase + wn * 64 + j * 16 + col16;
            v8h lo = *(const v8h*)&Lb[(size_t)rown * 32 + khalf];
            v8h hi = *(const v8h*)&Lb[(size_t)rown * 32 + khalf + 16];
            #pragma unroll
            for (int e = 0; e < 8; ++e) { lf[j][e] = lo[e]; lf[j][8 + e] = hi[e]; }
        }
        #pragma unroll
        for (int i = 0; i < 2; ++i)
            #pragma unroll
            for (int j = 0; j < 4; ++j)
                acc[i][j] = __builtin_amdgcn_wmma_f32_16x16x32_f16(
                    false, tf[i], false, lf[j], (short)0, acc[i][j], false, false);
    }

    // ---- bias + store (C layout: VGPR e -> M = e + 8*(lane>=16), N = lane&15) ----
    const int rofs = (lane >> 4) * 8;
    #pragma unroll
    for (int i = 0; i < 2; ++i) {
        #pragma unroll
        for (int j = 0; j < 4; ++j) {
            int m0   = mbase + wm * 32 + i * 16;
            int colN = nbase + wn * 64 + j * 16 + col16;
            float bv = bias[colN];
            #pragma unroll
            for (int e = 0; e < 8; ++e)
                C[(size_t)(m0 + rofs + e) * N + colN] = acc[i][j][e] + bv;
        }
    }
}

// ---------------------------------------------------------------------------
// Host launcher
// ---------------------------------------------------------------------------
extern "C" void kernel_launch(void* const* d_in, const int* in_sizes, int n_in,
                              void* d_out, int out_size, void* d_ws, size_t ws_size,
                              hipStream_t stream) {
    const float* x    = (const float*)d_in[0];  // [4,2048,4096]
    const float* sgn  = (const float*)d_in[1];  // [4096]
    // d_in[2] = H_block: deterministic Sylvester Hadamard -> butterfly instead
    const float* w    = (const float*)d_in[3];  // [4096,4096]
    const float* la   = (const float*)d_in[4];  // [32,4096]
    const float* lb   = (const float*)d_in[5];  // [4096,32]
    const float* bias = (const float*)d_in[6];  // [4096]
    float* out = (float*)d_out;

    char* ws = (char*)d_ws;
    size_t off = 0;
    _Float16* xq  = (_Float16*)(ws + off); off += (size_t)M_TOK * DK * sizeof(_Float16);   // 64 MB
    _Float16* w16 = (_Float16*)(ws + off); off += (size_t)DN * DK * sizeof(_Float16);      // 32 MB
    _Float16* lb16= (_Float16*)(ws + off); off += (size_t)DN * RANK * sizeof(_Float16);    // 256 KB
    _Float16* tm  = (_Float16*)(ws + off);                                                 // 512 KB

    prep_kernel<<<dim3(M_TOK), dim3(256), 0, stream>>>(x, sgn, la, xq, tm);
    convert_kernel<<<dim3(2048), dim3(256), 0, stream>>>(w, lb, w16, lb16,
                                                         DN * DK, DN * RANK);
    gemm_kernel<<<dim3(DN / 128, M_TOK / 128), dim3(256), 0, stream>>>(
        xq, w16, tm, lb16, bias, out, M_TOK, DN, DK);
}
